// PCM_42597485641967
// MI455X (gfx1250) — matrix operand order repaired and checked by reading, not compiled, not run on hardware
//
#include <hip/hip_runtime.h>
#include <hip/hip_bf16.h>
#include <math.h>

typedef __attribute__((ext_vector_type(16))) _Float16 v16h;
typedef __attribute__((ext_vector_type(8)))  _Float16 v8h;
typedef __attribute__((ext_vector_type(8)))  float    v8f;

#define DD 32
#define HH 32
#define WWID 32
#define NN (DD*HH*WWID)     // 32768 nodes
#define BB 2
#define CF 256              // feature channels
#define CC 21               // cam channels
#define FD 64               // feat_dim
#define PENC 48             // positional-encoding dim
#define KNBR 19             // stencil size
#define OUTC 128            // [theta | phi] fused output columns
#define MT 64               // rows (nodes) per workgroup in WMMA gemm
#define ASTR (CF + 8)       // padded LDS row stride in halves (528 B: 16B-aligned, bank-skewed)
#define WSTRIDE 20          // padded stride for softmax weights

// connectivity-2 stencil in dx-major, dy, dz order (|dx|+|dy|+|dz| <= 2) — 19 offsets
__device__ const int OFF_TBL[KNBR][3] = {
  {-1,-1, 0},
  {-1, 0,-1},{-1, 0, 0},{-1, 0, 1},
  {-1, 1, 0},
  { 0,-1,-1},{ 0,-1, 0},{ 0,-1, 1},
  { 0, 0,-1},{ 0, 0, 0},{ 0, 0, 1},
  { 0, 1,-1},{ 0, 1, 0},{ 0, 1, 1},
  { 1,-1, 0},
  { 1, 0,-1},{ 1, 0, 0},{ 1, 0, 1},
  { 1, 1, 0}};

__device__ __forceinline__ void node_coords(int n, int& d, int& h, int& w) {
  d = n >> 10; h = (n >> 5) & 31; w = n & 31;
}

__device__ __forceinline__ int nbr_of(int n, int k, bool& valid) {
  int d, h, w; node_coords(n, d, h, w);
  int nd = d + OFF_TBL[k][0];
  int nh = h + OFF_TBL[k][1];
  int nw = w + OFF_TBL[k][2];
  valid = (nd >= 0) && (nd < DD) && (nh >= 0) && (nh < HH) && (nw >= 0) && (nw < WWID);
  nd = min(max(nd, 0), DD - 1);
  nh = min(max(nh, 0), HH - 1);
  nw = min(max(nw, 0), WWID - 1);
  return (nd << 10) | (nh << 5) | nw;
}

// ---------------------------------------------------------------------------
// Geo path: compute pe[n, 48] analytically, then G[n, 0:64]=pe@geo_theta_w+b,
// G[n, 64:128]=pe@geo_phi_w+b.  One block (128 threads) per node.
// ---------------------------------------------------------------------------
__global__ void __launch_bounds__(OUTC) geo_gemm_kernel(
    const float* __restrict__ gtw, const float* __restrict__ gtb,
    const float* __restrict__ gpw, const float* __restrict__ gpb,
    float* __restrict__ G) {
  __shared__ float pe[PENC];
  const int n = blockIdx.x;
  const int t = threadIdx.x;          // 0..127
  if (t < PENC) {
    const int axis = t >> 4, rem = t & 15, j = rem >> 1, s = rem & 1;
    int d, h, w; node_coords(n, d, h, w);
    const int coord = (axis == 0) ? d : ((axis == 1) ? h : w);
    const float ang = (float)coord * powf(1e-4f, (float)j * 0.125f);
    pe[t] = s ? cosf(ang) : sinf(ang);
  }
  __syncthreads();
  float acc = (t < FD) ? gtb[t] : gpb[t - FD];
  const float* wcol = (t < FD) ? (gtw + t) : (gpw + (t - FD));
  #pragma unroll 8
  for (int p = 0; p < PENC; ++p) acc += pe[p] * wcol[(size_t)p * FD];
  G[(size_t)n * OUTC + t] = acc;
}

// ---------------------------------------------------------------------------
// Main WMMA GEMM: TP[b, n, 0:64] = f16(f_flat) @ theta_w + theta_b
//                 TP[b, n, 64:128] = f16(f_flat) @ phi_w  + phi_b
// f is [B, 256, N] (n contiguous): stage a transposed 64x256 f16 A-tile AND
// the full 256x128 fused weight panel (B-layout, converted once per block)
// in dynamic LDS (~99 KB; gfx1250 WGP has 320 KB).  8 waves; each wave owns
// a 16-column slice and FOUR 16x16 row subtiles (MT=64), so every B fragment
// (2x ds_load_b128) feeds 4 v_wmma_f32_16x16x32_f16.  K=256 fully unrolled:
// 32 WMMA per wave.
// ---------------------------------------------------------------------------
__global__ void __launch_bounds__(256) tp_gemm_kernel(
    const float* __restrict__ f,
    const float* __restrict__ thw, const float* __restrict__ thb,
    const float* __restrict__ phw, const float* __restrict__ phb,
    float* __restrict__ TP) {
  extern __shared__ __align__(16) _Float16 smem[];
  _Float16* As = smem;                 // [MT][ASTR]
  _Float16* Bs = smem + MT * ASTR;     // [OUTC][ASTR]

  const int tid  = threadIdx.x;
  const int nblk = NN / MT;                   // 512 node tiles
  const int bi   = blockIdx.x / nblk;
  const int n0   = (blockIdx.x % nblk) * MT;

  // ---- Stage A tile: f[bi][c][n0+n] -> As[n][c], f32 -> f16 (coalesced in n) ----
  {
    const int nloc  = tid & 63;               // lanes sweep contiguous n
    const int cbase = (tid >> 6) * 64;        // 4 thread-groups cover c
    const float* src = f + (size_t)bi * CF * NN + n0 + nloc;
    #pragma unroll
    for (int ci = 0; ci < 64; ++ci) {
      const int c = cbase + ci;
      if (ci + 8 < 64) __builtin_prefetch(src + (size_t)(c + 8) * NN, 0, 0);  // global_prefetch_b8
      As[nloc * ASTR + c] = (_Float16)src[(size_t)c * NN];
    }
  }
  // ---- Stage fused weight panel: Bs[j][c] = (j<64 ? theta_w : phi_w)[c][j'] ----
  {
    const int j  = tid & 127;                 // lanes sweep contiguous j (coalesced)
    const int cb = (tid >> 7) * 128;          // 2 thread-groups cover c
    const float* wc = (j < FD) ? (thw + j) : (phw + (j - FD));
    #pragma unroll
    for (int ci = 0; ci < 128; ++ci) {
      const int c = cb + ci;
      Bs[j * ASTR + c] = (_Float16)wc[(size_t)c * FD];
    }
  }
  __syncthreads();

  const int wave = tid >> 5;
  const int lane = tid & 31;
  const int hi16 = (lane >= 16) ? 1 : 0;
  const int jcol = wave * 16 + (lane & 15);   // output column
  const int m0   = lane & 15;                 // A-matrix row within subtile
  const int ksel = hi16 ? 16 : 0;             // B layout: lanes 16-31 hold K+16
  const int asel = hi16 ? 16 : 0;             // A layout: lanes 16-31 hold K+16

  v8f acc[4] = {{}, {}, {}, {}};
  #pragma unroll
  for (int kk = 0; kk < CF; kk += 32) {
    // B fragment (32x16 f16): half h -> K = kk + ksel + h, N = lane%16
    v16h bf;
    {
      const int koff = jcol * ASTR + kk + ksel;
      v8h b0 = *(const v8h*)&Bs[koff];
      v8h b1 = *(const v8h*)&Bs[koff + 8];
      #pragma unroll
      for (int i = 0; i < 8; ++i) { bf[i] = b0[i]; bf[8 + i] = b1[i]; }
    }
    // Four A fragments (16x32 f16 each) share this B fragment
    #pragma unroll
    for (int s = 0; s < 4; ++s) {
      v16h a;
      const int abase = (m0 + s * 16) * ASTR + kk + asel;
      v8h lo = *(const v8h*)&As[abase];
      v8h hi = *(const v8h*)&As[abase + 8];
      #pragma unroll
      for (int i = 0; i < 8; ++i) { a[i] = lo[i]; a[8 + i] = hi[i]; }
      acc[s] = __builtin_amdgcn_wmma_f32_16x16x32_f16(false, a, false, bf,
                                                      (short)0, acc[s], false, false);
    }
  }

  // ---- bias + store (C/D layout: lane<16 -> M=r, lane>=16 -> M=r+8) ----
  const float bias  = (jcol < FD) ? thb[jcol] : phb[jcol - FD];
  const int   rbase = hi16 ? 8 : 0;
  #pragma unroll
  for (int s = 0; s < 4; ++s) {
    #pragma unroll
    for (int r = 0; r < 8; ++r) {
      const int n = n0 + s * 16 + r + rbase;
      TP[((size_t)(bi * NN + n)) * OUTC + jcol] = acc[s][r] + bias;
    }
  }
}

// ---------------------------------------------------------------------------
// Scores + softmax: one wave32 per (b, n); lane k < 19 handles neighbor k.
// score = <theta_n, phi_nbr> + <g_theta_n, g_phi_nbr>, scaled by 1/sqrt(19).
// Wave-wide shuffle max/sum softmax (wave32-native).
// ---------------------------------------------------------------------------
__global__ void __launch_bounds__(256) score_kernel(
    const float* __restrict__ TP, const float* __restrict__ G,
    float* __restrict__ Wout) {
  const int wave = threadIdx.x >> 5;
  const int lane = threadIdx.x & 31;
  const int g = blockIdx.x * 8 + wave;        // 0 .. B*N-1
  const int bi = g / NN, n = g % NN;

  bool valid = false;
  float s = -1e30f;
  if (lane < KNBR) {
    const int nbr = nbr_of(n, lane, valid);
    if (valid) {
      const float* tn = TP + ((size_t)(bi * NN + n))   * OUTC;
      const float* tm = TP + ((size_t)(bi * NN + nbr)) * OUTC + FD;
      const float* gn = G + (size_t)n   * OUTC;
      const float* gm = G + (size_t)nbr * OUTC + FD;
      float acc = 0.f;
      #pragma unroll 8
      for (int d = 0; d < FD; ++d) acc += tn[d] * tm[d] + gn[d] * gm[d];
      s = acc * 0.22941573387056174f;         // 1/sqrt(19)
    }
  }
  float m = s;
  #pragma unroll
  for (int o = 16; o > 0; o >>= 1) m = fmaxf(m, __shfl_xor(m, o, 32));
  const float e = (lane < KNBR && valid) ? expf(s - m) : 0.f;
  float sum = e;
  #pragma unroll
  for (int o = 16; o > 0; o >>= 1) sum += __shfl_xor(sum, o, 32);
  if (lane < KNBR) Wout[(size_t)g * WSTRIDE + lane] = e / sum;
}

// ---------------------------------------------------------------------------
// Neighborhood aggregation: out[b,n,c] = sum_k w[b,n,k] * in[b, nbr(n,k), c]
// Layout flags pick channel-major ([B,C,N], matches cam/d_out) vs node-major
// ([B,N,C], internal ping buffer).  Invalid neighbors carry weight 0.
// ---------------------------------------------------------------------------
__global__ void __launch_bounds__(256) agg_kernel(
    const float* __restrict__ src, const float* __restrict__ Wk,
    float* __restrict__ dst, int srcNodeMajor, int dstNodeMajor) {
  const int g = blockIdx.x * blockDim.x + threadIdx.x;
  if (g >= BB * NN) return;
  const int bi = g / NN, n = g % NN;

  float wv[KNBR];
  const float* wp = Wk + (size_t)g * WSTRIDE;
  #pragma unroll
  for (int k = 0; k < KNBR; ++k) wv[k] = wp[k];

  float acc[CC];
  #pragma unroll
  for (int c = 0; c < CC; ++c) acc[c] = 0.f;

  for (int k = 0; k < KNBR; ++k) {
    bool valid; const int nbr = nbr_of(n, k, valid); (void)valid;
    const float wk = wv[k];
    if (srcNodeMajor) {
      const float* sp = src + ((size_t)(bi * NN + nbr)) * CC;
      #pragma unroll
      for (int c = 0; c < CC; ++c) acc[c] += wk * sp[c];
    } else {
      const float* sp = src + (size_t)bi * CC * NN + nbr;
      #pragma unroll
      for (int c = 0; c < CC; ++c) acc[c] += wk * sp[(size_t)c * NN];
    }
  }
  if (dstNodeMajor) {
    float* dp = dst + ((size_t)(bi * NN + n)) * CC;
    #pragma unroll
    for (int c = 0; c < CC; ++c) dp[c] = acc[c];
  } else {
    float* dp = dst + (size_t)bi * CC * NN + n;
    #pragma unroll
    for (int c = 0; c < CC; ++c) dp[(size_t)c * NN] = acc[c];
  }
}

// ---------------------------------------------------------------------------
extern "C" void kernel_launch(void* const* d_in, const int* in_sizes, int n_in,
                              void* d_out, int out_size, void* d_ws, size_t ws_size,
                              hipStream_t stream) {
  (void)in_sizes; (void)n_in; (void)out_size; (void)ws_size;
  const float* cam = (const float*)d_in[0];
  const float* f   = (const float*)d_in[1];
  const float* thw = (const float*)d_in[2];
  const float* thb = (const float*)d_in[3];
  const float* phw = (const float*)d_in[4];
  const float* phb = (const float*)d_in[5];
  const float* gtw = (const float*)d_in[6];
  const float* gtb = (const float*)d_in[7];
  const float* gpw = (const float*)d_in[8];
  const float* gpb = (const float*)d_in[9];
  // d_in[10] neighbor_idx / d_in[11] nbr_mask: recomputed on device from stencil.

  char* ws = (char*)d_ws;
  const size_t TP_BYTES = (size_t)BB * NN * OUTC * 4;   // 32 MB
  const size_t G_BYTES  = (size_t)NN * OUTC * 4;        // 16 MB
  const size_t W_BYTES  = (size_t)BB * NN * WSTRIDE * 4;
  float* TP   = (float*)(ws);
  float* G    = (float*)(ws + TP_BYTES);
  float* Wk   = (float*)(ws + TP_BYTES + G_BYTES);
  float* camA = (float*)(ws + TP_BYTES + G_BYTES + W_BYTES);
  float* out  = (float*)d_out;

  const size_t gemm_lds = (size_t)(MT + OUTC) * ASTR * sizeof(_Float16);  // ~99 KB

  hipLaunchKernelGGL(geo_gemm_kernel, dim3(NN), dim3(OUTC), 0, stream,
                     gtw, gtb, gpw, gpb, G);
  hipLaunchKernelGGL(tp_gemm_kernel, dim3(BB * NN / MT), dim3(256), gemm_lds, stream,
                     f, thw, thb, phw, phb, TP);
  hipLaunchKernelGGL(score_kernel, dim3(BB * NN / 8), dim3(256), 0, stream,
                     TP, G, Wk);
  hipLaunchKernelGGL(agg_kernel, dim3((BB * NN + 255) / 256), dim3(256), 0, stream,
                     cam, Wk, camA, 0, 1);   // iter 1: cam [B,C,N] -> camA [B,N,C]
  hipLaunchKernelGGL(agg_kernel, dim3((BB * NN + 255) / 256), dim3(256), 0, stream,
                     camA, Wk, out, 1, 0);   // iter 2: camA [B,N,C] -> out [B,C,N]
}
